// AttentionHead_18708877541474
// MI455X (gfx1250) — matrix operand order repaired and checked by reading, not compiled, not run on hardware
//
#include <hip/hip_runtime.h>
#include <hip/hip_bf16.h>

typedef __bf16 bf16_t;
typedef __attribute__((ext_vector_type(16))) __bf16 v16bf;
typedef __attribute__((ext_vector_type(8)))  float  v8f;
typedef __attribute__((ext_vector_type(4)))  unsigned int u32x4;
typedef __attribute__((ext_vector_type(8)))  int i32x8;
typedef __attribute__((ext_vector_type(4)))  int i32x4;

#define T_SEQ  2048
#define E_DIM  1024
#define H_DIM  64
#define NBATCH 8
#define SCALE  0.125f   // 1/sqrt(64)

// ---------- helpers ----------

static __device__ __forceinline__ v8f wmma_bf16(v16bf a, v16bf b, v8f c) {
  // D = A(16x32 bf16) * B(32x16 bf16) + C(16x16 f32)
  return __builtin_amdgcn_wmma_f32_16x16x32_bf16(false, a, false, b,
                                                 (short)0, c, false, false);
}

static __device__ __forceinline__ v16bf load16bf(const bf16_t* p0, const bf16_t* p1) {
  union { uint4 u[2]; v16bf v; } U;
  U.u[0] = *reinterpret_cast<const uint4*>(p0);
  U.u[1] = *reinterpret_cast<const uint4*>(p1);
  return U.v;
}

static __device__ __forceinline__ unsigned pk2(float a, float b) {
  union { bf16_t h; unsigned short s; } A, B;
  A.h = (bf16_t)a; B.h = (bf16_t)b;
  return (unsigned)A.s | ((unsigned)B.s << 16);
}

// ---------- kernel 1: swizzle W into B-fragment layout ----------
// Wswz[p][ks(32)][nt(4)][lane(32)][16] bf16,
// element i = W[ks*32 + (lane>=16?8:0) + i + (i>=8?8:0)][nt*16 + (lane&15)]
__global__ __launch_bounds__(256)
void prep_w_kernel(const float* __restrict__ Wq, const float* __restrict__ Wk,
                   const float* __restrict__ Wv, bf16_t* __restrict__ Wswz) {
  int tid = blockIdx.x * 256 + threadIdx.x;
  if (tid >= 3 * 32 * 4 * 32) return;
  int p  = tid / 4096;
  int r  = tid % 4096;
  int ks = r / 128;
  int r2 = r % 128;
  int nt = r2 / 32;
  int lane = r2 % 32;
  const float* W = (p == 0) ? Wq : ((p == 1) ? Wk : Wv);
  int colh = nt * 16 + (lane & 15);
  int kb = ks * 32 + ((lane >= 16) ? 8 : 0);
  union { bf16_t e[16]; uint4 u[2]; } U;
#pragma unroll
  for (int i = 0; i < 16; ++i)
    U.e[i] = (bf16_t)W[(size_t)(kb + i + (i >= 8 ? 8 : 0)) * H_DIM + colh];
  uint4* dst = reinterpret_cast<uint4*>(Wswz + (size_t)tid * 16);
  dst[0] = U.u[0];
  dst[1] = U.u[1];
}

// ---------- kernel 2: Q/K/V projections via WMMA + TDM-staged W in LDS ----------
// grid = (row_blocks, 3). blockIdx.y selects projection. One WG = 8 waves,
// wave owns 16 rows x 64 cols. W slab (128 KB, pre-swizzled) is DMA'd to LDS
// once per WG by the Tensor Data Mover, then fragments come from ds_load_b128.
__global__ __launch_bounds__(256)
void proj_kernel(const float* __restrict__ X, const bf16_t* __restrict__ Wswz,
                 const float* __restrict__ bq, const float* __restrict__ bk,
                 const float* __restrict__ bv,
                 bf16_t* __restrict__ Qb, bf16_t* __restrict__ Kb,
                 bf16_t* __restrict__ Vb) {
  __shared__ bf16_t w_lds[32 * 4 * 32 * 16];   // 65536 bf16 = 128 KB

  int wave = threadIdx.x >> 5, lane = threadIdx.x & 31;
  int col = lane & 15;
  int hi  = (lane >> 4) & 1;
  int p   = blockIdx.y;
  int row0 = blockIdx.x * 128 + wave * 16;

  // Dead (never-taken, unprovable) store: the TDM write below is invisible to
  // alias analysis, and without any visible store the compiler folds all reads
  // of w_lds to undef. This makes w_lds a "written" object.
  if (Wswz == nullptr) w_lds[threadIdx.x] = (bf16_t)0.0f;

  // --- TDM: one descriptor copies this projection's 128 KB W slab to LDS ---
  if (wave == 0) {
    unsigned long long ga =
        (unsigned long long)(uintptr_t)(Wswz + (size_t)p * 65536);
    u32x4 g0; i32x8 g1; i32x4 g2; i32x4 g3; i32x8 g4;
    // D# group0: count=1, lds_addr=0 (sole LDS array), global_addr, type=2
    g0[0] = 1u;
    g0[1] = 0u;
    g0[2] = (unsigned)(ga & 0xffffffffull);
    g0[3] = (unsigned)((ga >> 32) & 0x01ffffffull) | (2u << 30);
    // D# group1: data_size=3 (8B), tensor_dim0=16384, tensor_dim1=1,
    //            tile_dim0=16384, tile_dim1=1, stride0=16384
    g1[0] = (int)(3u << 16);
    g1[1] = (int)0x40000000;   // tensor_dim0[15:0]=16384 in bits 31:16
    g1[2] = (int)0x00010000;   // tensor_dim0[31:16]=0, tensor_dim1[15:0]=1
    g1[3] = (int)0x40000000;   // tensor_dim1[31:16]=0, tile_dim0=16384
    g1[4] = 1;                 // tile_dim1=1, tile_dim2=0
    g1[5] = 16384;             // tensor_dim0_stride[31:0]
    g1[6] = 0;
    g1[7] = 0;
    g2[0] = 0; g2[1] = 0; g2[2] = 0; g2[3] = 0;
    g3[0] = 0; g3[1] = 0; g3[2] = 0; g3[3] = 0;
    g4[0] = 0; g4[1] = 0; g4[2] = 0; g4[3] = 0;
    g4[4] = 0; g4[5] = 0; g4[6] = 0; g4[7] = 0;
    __builtin_amdgcn_tensor_load_to_lds(g0, g1, g2, g3, g4, 0);
    __builtin_amdgcn_s_wait_tensorcnt(0);
  }
  // opaque writer: w_lds is an addrspace(3) global, so a memory-clobber asm is
  // a potential store to it -> forces real ds_loads after the barrier.
  asm volatile("" ::: "memory");
  __syncthreads();
  asm volatile("" ::: "memory");

  v8f zero8 = {0.f,0.f,0.f,0.f,0.f,0.f,0.f,0.f};
  v8f acc[4] = {zero8, zero8, zero8, zero8};

  const float* xrow = X + (size_t)(row0 + col) * E_DIM;

  for (int ks = 0; ks < 32; ++ks) {
    // A fragment: lane<16 holds K {0..7,16..23}, lane>=16 holds {8..15,24..31}
    const float* xp = xrow + ks * 32 + hi * 8;
    float4 c0a = *reinterpret_cast<const float4*>(xp);
    float4 c0b = *reinterpret_cast<const float4*>(xp + 4);
    float4 c1a = *reinterpret_cast<const float4*>(xp + 16);
    float4 c1b = *reinterpret_cast<const float4*>(xp + 20);
    v16bf af;
    af[0]=(bf16_t)c0a.x; af[1]=(bf16_t)c0a.y; af[2]=(bf16_t)c0a.z; af[3]=(bf16_t)c0a.w;
    af[4]=(bf16_t)c0b.x; af[5]=(bf16_t)c0b.y; af[6]=(bf16_t)c0b.z; af[7]=(bf16_t)c0b.w;
    af[8]=(bf16_t)c1a.x; af[9]=(bf16_t)c1a.y; af[10]=(bf16_t)c1a.z; af[11]=(bf16_t)c1a.w;
    af[12]=(bf16_t)c1b.x; af[13]=(bf16_t)c1b.y; af[14]=(bf16_t)c1b.z; af[15]=(bf16_t)c1b.w;

#pragma unroll
    for (int nt = 0; nt < 4; ++nt) {
      const bf16_t* wp = &w_lds[((ks * 4 + nt) * 32 + lane) * 16];
      acc[nt] = wmma_bf16(af, load16bf(wp, wp + 8), acc[nt]);
    }
  }

  const float* bias = (p == 0) ? bq : ((p == 1) ? bk : bv);
  bf16_t* outp      = (p == 0) ? Qb : ((p == 1) ? Kb : Vb);
#pragma unroll
  for (int nt = 0; nt < 4; ++nt) {
    float bb = bias[nt * 16 + col];
#pragma unroll
    for (int r = 0; r < 8; ++r) {
      int row = row0 + hi * 8 + r;
      outp[(size_t)row * H_DIM + nt * 16 + col] = (bf16_t)(acc[nt][r] + bb);
    }
  }
}

// ---------- kernel 3: V -> V^T (per batch) ----------
__global__ __launch_bounds__(256)
void vtrans_kernel(const bf16_t* __restrict__ Vb, bf16_t* __restrict__ Vt) {
  size_t i = (size_t)blockIdx.x * 256 + threadIdx.x;   // B*H*T = 1M threads
  int t = (int)(i & (T_SEQ - 1));
  int rest = (int)(i >> 11);
  int h = rest & (H_DIM - 1);
  int b = rest >> 6;
  Vt[((size_t)b * H_DIM + h) * T_SEQ + t] = Vb[((size_t)b * T_SEQ + t) * H_DIM + h];
}

// ---------- kernel 4: transposed flash attention ----------
// Wave owns 16 queries. S^T = K * Q^T, O^T = V^T * P^T; lane = one query column.
__global__ __launch_bounds__(256)
void attn_kernel(const bf16_t* __restrict__ Qb, const bf16_t* __restrict__ Kb,
                 const bf16_t* __restrict__ Vt, float* __restrict__ out) {
  int wave = threadIdx.x >> 5, lane = threadIdx.x & 31;
  int col = lane & 15;
  int hi  = (lane >> 4) & 1;
  int qblk = blockIdx.x & 15;
  int b    = blockIdx.x >> 4;
  int q0 = qblk * 128 + wave * 16;
  int q  = q0 + col;

  const bf16_t* Qp = Qb + (size_t)b * T_SEQ * H_DIM;
  const bf16_t* Kp = Kb + (size_t)b * T_SEQ * H_DIM;
  const bf16_t* Vp = Vt + (size_t)b * H_DIM * T_SEQ;

  // Q^T B-fragments (kept in regs): lane holds column q, h-range per half-wave
  v16bf qf[2];
#pragma unroll
  for (int c = 0; c < 2; ++c) {
    const bf16_t* p = Qp + (size_t)q * H_DIM + c * 32 + hi * 8;
    qf[c] = load16bf(p, p + 16);
  }

  v8f zero8 = {0.f,0.f,0.f,0.f,0.f,0.f,0.f,0.f};
  v8f acc[4] = {zero8, zero8, zero8, zero8};   // O^T, 64 h-rows x 16 q-cols
  float m = -__builtin_inff();
  float l = 0.f;

  int jpmax = (q0 + 15) >> 5;                  // 32-key pairs
  for (int jp = 0; jp <= jpmax; ++jp) {
    int jb = jp * 32;

    if (jp < jpmax) {  // cover next iteration's latency
      __builtin_prefetch(Kp + (size_t)(jb + 32 + col) * H_DIM, 0, 0);
      __builtin_prefetch(Vp + (size_t)col * T_SEQ + jb + 32, 0, 0);
    }

    v8f s0c = zero8, s1c = zero8;
#pragma unroll
    for (int c = 0; c < 2; ++c) {              // h in steps of 32
      const bf16_t* kp0 = Kp + (size_t)(jb + col) * H_DIM + c * 32 + hi * 8;
      s0c = wmma_bf16(load16bf(kp0, kp0 + 16), qf[c], s0c);
      const bf16_t* kp1 = Kp + (size_t)(jb + 16 + col) * H_DIM + c * 32 + hi * 8;
      s1c = wmma_bf16(load16bf(kp1, kp1 + 16), qf[c], s1c);
    }

    // scale + causal mask (key index is per-VGPR-row)
    float s0[8], s1[8];
#pragma unroll
    for (int r = 0; r < 8; ++r) {
      int k0 = jb + hi * 8 + r;
      float v0 = s0c[r] * SCALE;
      s0[r] = (k0 <= q) ? v0 : -__builtin_inff();
      float v1 = s1c[r] * SCALE;
      s1[r] = (k0 + 16 <= q) ? v1 : -__builtin_inff();
    }

    // online softmax: per-lane (one query) + one cross-half shuffle
    float mloc = s0[0];
#pragma unroll
    for (int r = 1; r < 8; ++r) mloc = fmaxf(mloc, s0[r]);
#pragma unroll
    for (int r = 0; r < 8; ++r) mloc = fmaxf(mloc, s1[r]);
    mloc = fmaxf(mloc, __shfl_xor(mloc, 16, 32));
    float mnew = fmaxf(m, mloc);
    float alpha = __expf(m - mnew);

    float p0v[8], p1v[8], sum = 0.f;
#pragma unroll
    for (int r = 0; r < 8; ++r) {
      p0v[r] = __expf(s0[r] - mnew);
      p1v[r] = __expf(s1[r] - mnew);
      sum += p0v[r] + p1v[r];
    }
    sum += __shfl_xor(sum, 16, 32);
    l = l * alpha + sum;
    m = mnew;
#pragma unroll
    for (int mt = 0; mt < 4; ++mt)
#pragma unroll
      for (int r = 0; r < 8; ++r) acc[mt][r] *= alpha;

    // pack P^T into B-fragment layout: lane<16 needs keys{0..15}, lane>=16 keys{16..31}
    unsigned pj[4], pj1[4];
#pragma unroll
    for (int i = 0; i < 4; ++i) {
      pj[i]  = pk2(p0v[2 * i], p0v[2 * i + 1]);
      pj1[i] = pk2(p1v[2 * i], p1v[2 * i + 1]);
    }
    union { unsigned u[8]; v16bf v; } pf;
#pragma unroll
    for (int i = 0; i < 4; ++i) {
      unsigned oj  = (unsigned)__shfl_xor((int)pj[i], 16, 32);
      unsigned oj1 = (unsigned)__shfl_xor((int)pj1[i], 16, 32);
      pf.u[i]     = hi ? oj1   : pj[i];   // keys {16..23} : {0..7}
      pf.u[4 + i] = hi ? pj1[i] : oj;     // keys {24..31} : {8..15}
    }

    // O^T += V^T * P^T  (4 h-tiles of 16)
#pragma unroll
    for (int mt = 0; mt < 4; ++mt) {
      const bf16_t* vp = Vp + (size_t)(mt * 16 + col) * T_SEQ + jb + hi * 16;
      acc[mt] = wmma_bf16(load16bf(vp, vp + 8), pf.v, acc[mt]);
    }
  }

  // normalize per-lane (one query each) and store O row-major
  float rcp = 1.0f / l;
#pragma unroll
  for (int mt = 0; mt < 4; ++mt) {
    float* op = out + ((size_t)b * T_SEQ + q) * H_DIM + mt * 16 + hi * 8;
    float4 o0 = make_float4(acc[mt][0]*rcp, acc[mt][1]*rcp, acc[mt][2]*rcp, acc[mt][3]*rcp);
    float4 o1 = make_float4(acc[mt][4]*rcp, acc[mt][5]*rcp, acc[mt][6]*rcp, acc[mt][7]*rcp);
    *reinterpret_cast<float4*>(op) = o0;
    *reinterpret_cast<float4*>(op + 4) = o1;
  }
}

// ---------- host launcher ----------
extern "C" void kernel_launch(void* const* d_in, const int* in_sizes, int n_in,
                              void* d_out, int out_size, void* d_ws, size_t ws_size,
                              hipStream_t stream) {
  (void)in_sizes; (void)n_in; (void)out_size; (void)ws_size;
  const float* X  = (const float*)d_in[0];
  const float* Wk = (const float*)d_in[1];
  const float* bk = (const float*)d_in[2];
  const float* Wq = (const float*)d_in[3];
  const float* bq = (const float*)d_in[4];
  const float* Wv = (const float*)d_in[5];
  const float* bv = (const float*)d_in[6];
  float* out = (float*)d_out;

  char* ws = (char*)d_ws;
  const size_t SZ_QKV = (size_t)NBATCH * T_SEQ * H_DIM * sizeof(bf16_t); // 2 MB
  bf16_t* Qb   = (bf16_t*)(ws);
  bf16_t* Kb   = (bf16_t*)(ws + SZ_QKV);
  bf16_t* Vb   = (bf16_t*)(ws + 2 * SZ_QKV);
  bf16_t* Vt   = (bf16_t*)(ws + 3 * SZ_QKV);
  bf16_t* Wswz = (bf16_t*)(ws + 4 * SZ_QKV);   // 384 KB

  prep_w_kernel<<<48, 256, 0, stream>>>(Wq, Wk, Wv, Wswz);
  dim3 pgrid((NBATCH * T_SEQ) / 128, 3);
  proj_kernel<<<pgrid, 256, 0, stream>>>(X, Wswz, bq, bk, bv, Qb, Kb, Vb);
  vtrans_kernel<<<((size_t)NBATCH * H_DIM * T_SEQ) / 256, 256, 0, stream>>>(Vb, Vt);
  attn_kernel<<<NBATCH * (T_SEQ / 128), 256, 0, stream>>>(Qb, Kb, Vt, out);
}
